// DAG_Generator_19224273617536
// MI455X (gfx1250) — compile-verified, matrix operand order checked by reading, not run on hardware
//
#include <hip/hip_runtime.h>

typedef __attribute__((ext_vector_type(16))) _Float16 v16h;
typedef __attribute__((ext_vector_type(8)))  float    v8f;

#define B_TOTAL 131072
#define IDIM    16
#define HDIM    64
#define INPAD   49

// ---- workspace layout (bytes) ----
#define WS_THETA_OFF 0        // 1024 f32 = 4096 B
#define WS_W1H_OFF   4096     // 65536 f16 = 131072 B (fragment layout)

// ---- LDS layout (bytes) ----
#define SM_W1_OFF   0                     // 131072 B : W1 f16 fragments
#define SM_WC_OFF   131072                // 10240 B  : Wc f32 (10x256)
#define SM_TH_OFF   141312                // 4096 B   : theta f32 (4x256)
#define SM_B1_OFF   145408                // 4096 B   : b1 f32 (16x64)
#define SM_W2_OFF   149504                // 4096 B   : W2 f32 (16x64)
#define SM_B2_OFF   153600                // 64 B     : b2 f32 (16)
#define SM_PART_OFF 153664                // 8192 B   : 8 waves x 16x16 f32
#define SM_TOTAL    161856

union HFrag {
    v16h v;
    _Float16 h[16];
    uint4 q[2];
};

// Prep: theta = mu + log1p(exp(sigma))*noise_d, and pack W1 into per-lane
// f16 B-matrix fragments: frag[(i*8 + s*4 + t)*32 + lane] = 16 halves.
// B 32x16 layout (symmetric to A): lane nl (0-15) holds column n=nl with
// K = {s*32 + 0..7, s*32 + 16..23}; lane nl+16 holds K = {+8..15, +24..31}.
__global__ void dag_prep_kernel(const float* __restrict__ noise_d,
                                const float* __restrict__ mu,
                                const float* __restrict__ sigma,
                                const float* __restrict__ W1,
                                float* __restrict__ theta_ws,
                                uint4* __restrict__ w1h_ws) {
    int tid = blockIdx.x * blockDim.x + threadIdx.x;   // 0..4095
    if (tid < 1024) {
        float s = sigma[tid];
        theta_ws[tid] = mu[tid] + log1pf(expf(s)) * noise_d[tid];
    }
    int i  = tid >> 8;
    int rest = tid & 255;
    int s  = rest >> 7;
    int t  = (rest >> 5) & 3;
    int L  = rest & 31;
    int kh = L >> 4;
    int nl = L & 15;
    int n  = t * 16 + nl;
    HFrag f;
#pragma unroll
    for (int j = 0; j < 16; ++j) {
        int kl = (j < 8) ? (kh * 8 + j) : (16 + kh * 8 + (j - 8));
        int k  = s * 32 + kl;
        float val = (k < INPAD) ? W1[(i * INPAD + k) * HDIM + n] : 0.0f;
        f.h[j] = (_Float16)val;
    }
    w1h_ws[tid * 2 + 0] = f.q[0];
    w1h_ws[tid * 2 + 1] = f.q[1];
}

__global__ void dag_main_kernel(const float* __restrict__ noise,
                                const float* __restrict__ inC,
                                const float* __restrict__ inD,
                                const float* __restrict__ Wc,
                                const float* __restrict__ b1,
                                const float* __restrict__ W2,
                                const float* __restrict__ b2,
                                const float* __restrict__ theta_ws,
                                const uint4* __restrict__ w1h_ws,
                                float* __restrict__ out) {
    extern __shared__ __align__(16) char smem[];
    uint4* s_w1  = (uint4*)(smem + SM_W1_OFF);      // 8192 uint4
    float* s_wc  = (float*)(smem + SM_WC_OFF);
    float* s_th  = (float*)(smem + SM_TH_OFF);
    float* s_b1  = (float*)(smem + SM_B1_OFF);
    float* s_w2  = (float*)(smem + SM_W2_OFF);
    float* s_b2  = (float*)(smem + SM_B2_OFF);
    float* s_part = (float*)(smem + SM_PART_OFF);

    const int tid = threadIdx.x;
    // ---- cooperative staging (256 threads) ----
#pragma unroll
    for (int r = 0; r < 32; ++r) s_w1[tid + r * 256] = w1h_ws[tid + r * 256];
#pragma unroll
    for (int r = 0; r < 10; ++r) s_wc[tid + r * 256] = Wc[tid + r * 256];
#pragma unroll
    for (int r = 0; r < 4; ++r) {
        s_th[tid + r * 256] = theta_ws[tid + r * 256];
        s_b1[tid + r * 256] = b1[tid + r * 256];
        s_w2[tid + r * 256] = W2[tid + r * 256];
    }
    if (tid < 16) s_b2[tid] = b2[tid];
    __syncthreads();

    const int wid  = tid >> 5;
    const int lane = tid & 31;
    const int nl   = lane & 15;
    const int kh   = lane >> 4;       // K-half: 0 -> K 0..7/16..23, 1 -> K 8..15/24..31
    const int tile = blockIdx.x * 8 + wid;
    const int row  = tile * 16 + nl;  // A-matrix row M = nl for both half-lane groups

    float c10[10];
#pragma unroll
    for (int t = 0; t < 10; ++t) c10[t] = inC[row * 10 + t];
    const float4 d4 = ((const float4*)inD)[row];
    const float dv[4] = {d4.x, d4.y, d4.z, d4.w};

    float* part = s_part + wid * 256;
    float pval = 0.0f;                // parent scalar for this lane's row

    for (int i = 0; i < IDIM; ++i) {
        const int cbase = i * 16 + kh * 8;

        // ci = input_c @ Wc slice, di = input_d @ theta slice (8 halves each)
        float cif[8], dif[8];
#pragma unroll
        for (int j = 0; j < 8; ++j) { cif[j] = 0.0f; dif[j] = 0.0f; }
#pragma unroll
        for (int t = 0; t < 10; ++t) {
            const float* wrow = s_wc + t * 256 + cbase;
#pragma unroll
            for (int j = 0; j < 8; ++j) cif[j] = fmaf(c10[t], wrow[j], cif[j]);
        }
#pragma unroll
        for (int t = 0; t < 4; ++t) {
            const float* trow = s_th + t * 256 + cbase;
#pragma unroll
            for (int j = 0; j < 8; ++j) dif[j] = fmaf(dv[t], trow[j], dif[j]);
        }

        // A fragment 0: K=0..31  -> [ci | di]
        HFrag a0;
#pragma unroll
        for (int j = 0; j < 8; ++j) {
            a0.h[j]     = (_Float16)cif[j];
            a0.h[8 + j] = (_Float16)dif[j];
        }
        // A fragment 1: K=32..63 -> [ni | parent,0...]
        const float4* np = (const float4*)(noise + (size_t)row * 256 + cbase);
        float4 n0 = np[0], n1 = np[1];
        HFrag a1;
        a1.h[0] = (_Float16)n0.x; a1.h[1] = (_Float16)n0.y;
        a1.h[2] = (_Float16)n0.z; a1.h[3] = (_Float16)n0.w;
        a1.h[4] = (_Float16)n1.x; a1.h[5] = (_Float16)n1.y;
        a1.h[6] = (_Float16)n1.z; a1.h[7] = (_Float16)n1.w;
#pragma unroll
        for (int j = 8; j < 16; ++j) a1.h[j] = (_Float16)0.0f;
        if (kh == 0) a1.h[8] = (_Float16)pval;   // K=48 = parent column

        // 8 WMMAs: 2 K-steps x 4 N-tiles, f32 accumulate
        v8f acc[4];
#pragma unroll
        for (int t = 0; t < 4; ++t) {
#pragma unroll
            for (int r = 0; r < 8; ++r) acc[t][r] = 0.0f;
        }
#pragma unroll
        for (int t = 0; t < 4; ++t) {
            HFrag bf0, bf1;
            int f0 = ((i * 8 + t) * 32 + lane) * 2;
            int f1 = ((i * 8 + 4 + t) * 32 + lane) * 2;
            bf0.q[0] = s_w1[f0]; bf0.q[1] = s_w1[f0 + 1];
            bf1.q[0] = s_w1[f1]; bf1.q[1] = s_w1[f1 + 1];
            acc[t] = __builtin_amdgcn_wmma_f32_16x16x32_f16(
                false, a0.v, false, bf0.v, (short)0, acc[t], false, false);
            acc[t] = __builtin_amdgcn_wmma_f32_16x16x32_f16(
                false, a1.v, false, bf1.v, (short)0, acc[t], false, false);
        }

        // epilogue: out_m = sum_n relu(h[m,n] + b1[n]) * W2[n]
        float psum[8];
#pragma unroll
        for (int r = 0; r < 8; ++r) psum[r] = 0.0f;
#pragma unroll
        for (int t = 0; t < 4; ++t) {
            float b1v = s_b1[i * 64 + t * 16 + nl];
            float w2v = s_w2[i * 64 + t * 16 + nl];
#pragma unroll
            for (int r = 0; r < 8; ++r) {
                float hh = acc[t][r] + b1v;
                hh = hh > 0.0f ? hh : 0.0f;
                psum[r] = fmaf(hh, w2v, psum[r]);
            }
        }
        // wave-local transpose-reduce through LDS: part[row][col]
        const int mbase = (lane < 16) ? 0 : 8;   // C layout: low lanes M=r, high lanes M=8+r
#pragma unroll
        for (int r = 0; r < 8; ++r) part[(mbase + r) * 16 + nl] = psum[r];
        asm volatile("s_wait_dscnt 0" ::: "memory");

        const float4* prow = (const float4*)(part + nl * 16);
        float4 q0 = prow[0], q1 = prow[1], q2 = prow[2], q3 = prow[3];
        float o = ((q0.x + q0.y) + (q0.z + q0.w)) + ((q1.x + q1.y) + (q1.z + q1.w))
                + ((q2.x + q2.y) + (q2.z + q2.w)) + ((q3.x + q3.y) + (q3.z + q3.w));
        o += s_b2[i];

        if (lane < 16) out[(size_t)row * 16 + i] = o;   // out[:, i]
        pval = (i <= 12) ? o : 0.0f;  // node i+1 has parent i exactly for i in 0..12
        asm volatile("s_wait_dscnt 0" ::: "memory");    // part WAR before next node
    }
}

extern "C" void kernel_launch(void* const* d_in, const int* in_sizes, int n_in,
                              void* d_out, int out_size, void* d_ws, size_t ws_size,
                              hipStream_t stream) {
    const float* noise   = (const float*)d_in[0];
    const float* input_c = (const float*)d_in[1];
    const float* input_d = (const float*)d_in[2];
    const float* noise_d = (const float*)d_in[3];
    const float* mu      = (const float*)d_in[4];
    const float* sigma   = (const float*)d_in[5];
    const float* Wc      = (const float*)d_in[6];
    const float* W1      = (const float*)d_in[7];
    const float* b1      = (const float*)d_in[8];
    const float* W2      = (const float*)d_in[9];
    const float* b2      = (const float*)d_in[10];
    float* out = (float*)d_out;

    float* theta_ws = (float*)((char*)d_ws + WS_THETA_OFF);
    uint4* w1h_ws   = (uint4*)((char*)d_ws + WS_W1H_OFF);

    dag_prep_kernel<<<16, 256, 0, stream>>>(noise_d, mu, sigma, W1, theta_ws, w1h_ws);

    const int tiles  = B_TOTAL / 16;        // 8192
    const int blocks = tiles / 8;           // 1024 (8 waves/block, 1 tile/wave)
    dag_main_kernel<<<blocks, 256, SM_TOTAL, stream>>>(
        noise, input_c, input_d, Wc, b1, W2, b2,
        theta_ws, (const uint4*)w1h_ws, out);
}
